// pcaWhitening_2534030704756
// MI455X (gfx1250) — matrix operand order repaired and checked by reading, not compiled, not run on hardware
//
#include <hip/hip_runtime.h>
#include <hip/hip_bf16.h>

// ---------------------------------------------------------------------------
// PCA-whitening-quantize pipeline for MI455X (gfx1250, wave32, WMMA).
//   B=128 C=64 H=W=56 MB=4  ->  D=1024, N=25088
// Pipeline: mean -> gather(im, imT bf16) -> cov GEMM (WMMA bf16->f32)
//           -> Jacobi eigensolver (fp32 VALU) -> proj GEMM -> quantize
//           -> back GEMM + NCHW scatter.
// GEMM wave tile = 64x32 (8 v8f accumulators) with SGPR-base + 32-bit
// per-lane offsets so nothing spills to scratch.
// ---------------------------------------------------------------------------

#define DD 1024
#define NN 25088
#define BB 128
#define CC 64
#define HH 56
#define SWEEPS 4

typedef __attribute__((ext_vector_type(16))) __bf16 v16bf;
typedef __attribute__((ext_vector_type(8)))  float  v8f;

union FragBF { uint4 u[2]; v16bf v; };

__device__ __forceinline__ unsigned monoKey(float f) {
  unsigned u = __float_as_uint(f);
  return (u & 0x80000000u) ? ~u : (u | 0x80000000u);
}
__device__ __forceinline__ float monoDec(unsigned k) {
  unsigned u = (k & 0x80000000u) ? (k ^ 0x80000000u) : ~k;
  return __uint_as_float(u);
}

// ---------------------------------------------------------------- mean ------
__global__ void mean_kernel(const float* __restrict__ x, float* __restrict__ mean) {
  __shared__ float red[256];
  const int d = blockIdx.x;
  const int c = d >> 4, mr = (d >> 2) & 3, mc = d & 3;
  float s = 0.f;
  for (int n = threadIdx.x; n < NN; n += blockDim.x) {
    const int b = n & 127, t2 = n >> 7;
    const int j = t2 % 14, i = t2 / 14;
    s += x[((b * CC + c) * HH + i * 4 + mr) * HH + j * 4 + mc];
  }
  red[threadIdx.x] = s;
  __syncthreads();
  for (int off = 128; off; off >>= 1) {
    if (threadIdx.x < off) red[threadIdx.x] += red[threadIdx.x + off];
    __syncthreads();
  }
  if (threadIdx.x == 0) mean[d] = red[0] * (1.0f / (float)NN);
}

// ------------------------------------------------------------- gather -------
// im[d][n] = x[...] - mean[d]  (n fastest -> coalesced im writes)
__global__ void gather_im(const float* __restrict__ x, const float* __restrict__ mean,
                          __bf16* __restrict__ im) {
  const int t = blockIdx.x * blockDim.x + threadIdx.x;
  const int n = t % NN, d = t / NN;
  const int b = n & 127, t2 = n >> 7, j = t2 % 14, i = t2 / 14;
  const int c = d >> 4, mr = (d >> 2) & 3, mc = d & 3;
  const float v = x[((b * CC + c) * HH + i * 4 + mr) * HH + j * 4 + mc] - mean[d];
  im[t] = (__bf16)v;
}
// imT[n][d] (d fastest -> coalesced imT writes)
__global__ void gather_imT(const float* __restrict__ x, const float* __restrict__ mean,
                           __bf16* __restrict__ imT) {
  const int t = blockIdx.x * blockDim.x + threadIdx.x;
  const int d = t & (DD - 1), n = t >> 10;
  const int b = n & 127, t2 = n >> 7, j = t2 % 14, i = t2 / 14;
  const int c = d >> 4, mr = (d >> 2) & 3, mc = d & 3;
  const float v = x[((b * CC + c) * HH + i * 4 + mr) * HH + j * 4 + mc] - mean[d];
  imT[t] = (__bf16)v;
}

// ---------------------------------------------------------- WMMA GEMM -------
// C[row, col] = sum_k A[row][k] * Bs[col][k]   (both row-major along k, so every
// operand load is per-lane contiguous : k = 16*(lane/16) + s in both A and B).
// Wave tile 64x32; block = 8 waves as 2(M) x 4(cols) -> 128x128 block tile.
// MODE 0: cov     -> fp32 store * (1/N)
// MODE 1: imProjT -> bf16 store + global max/min (monotone-key atomics)
// MODE 2: back    -> fp32 scatter to NCHW output
template <int MODE>
__global__ __launch_bounds__(256) void gemm_wmma(
    const __bf16* __restrict__ A, const __bf16* __restrict__ Bs,
    int lda, int ldb, int K, void* __restrict__ outp, int ldo,
    unsigned* __restrict__ mm) {
  const int wid  = threadIdx.x >> 5;
  const int lane = threadIdx.x & 31;
  const int lm   = lane & 15;
  const int lh   = lane >> 4;
  const int row0 = blockIdx.y * 128 + (wid & 1) * 64;   // 64 rows / wave
  const int col0 = blockIdx.x * 128 + (wid >> 1) * 32;  // 32 cols / wave

  // per-lane 32-bit element offsets (base pointers stay uniform in SGPRs)
  unsigned aoff[4], boff[2];
#pragma unroll
  for (int ti = 0; ti < 4; ++ti)
    aoff[ti] = (unsigned)(row0 + 16 * ti + lm) * (unsigned)lda + (unsigned)(lh * 16);
#pragma unroll
  for (int tj = 0; tj < 2; ++tj)
    boff[tj] = (unsigned)(col0 + 16 * tj + lm) * (unsigned)ldb + (unsigned)(lh * 16);

  v8f z = {};
  v8f acc[4][2];
#pragma unroll
  for (int i = 0; i < 4; ++i)
#pragma unroll
    for (int j = 0; j < 2; ++j) acc[i][j] = z;

  for (int k0 = 0; k0 < K; k0 += 32) {
    // prefetch 8 K-slabs (256 elements) ahead (emits global_prefetch_b8)
    __builtin_prefetch((const void*)(A + aoff[0] + 256), 0, 1);
    __builtin_prefetch((const void*)(Bs + boff[0] + 256), 0, 1);

    FragBF a[4], b[2];
#pragma unroll
    for (int ti = 0; ti < 4; ++ti) {
      a[ti].u[0] = *(const uint4*)(A + aoff[ti]);
      a[ti].u[1] = *(const uint4*)(A + aoff[ti] + 8);
      aoff[ti] += 32;
    }
#pragma unroll
    for (int tj = 0; tj < 2; ++tj) {
      b[tj].u[0] = *(const uint4*)(Bs + boff[tj]);
      b[tj].u[1] = *(const uint4*)(Bs + boff[tj] + 8);
      boff[tj] += 32;
    }
#pragma unroll
    for (int ti = 0; ti < 4; ++ti)
#pragma unroll
      for (int tj = 0; tj < 2; ++tj)
        acc[ti][tj] = __builtin_amdgcn_wmma_f32_16x16x32_bf16(
            false, a[ti].v, false, b[tj].v, (short)0, acc[ti][tj], false, false);
  }

  float lmax = -3.4e38f, lmin = 3.4e38f;
#pragma unroll
  for (int ti = 0; ti < 4; ++ti) {
#pragma unroll
    for (int tj = 0; tj < 2; ++tj) {
#pragma unroll
      for (int r = 0; r < 8; ++r) {
        const int row = row0 + 16 * ti + r + lh * 8;
        const int col = col0 + 16 * tj + lm;
        const float v = acc[ti][tj][r];
        if (MODE == 0) {
          ((float*)outp)[(size_t)row * ldo + col] = v * (1.0f / (float)NN);
        } else if (MODE == 1) {
          ((__bf16*)outp)[(size_t)row * ldo + col] = (__bf16)v;
          lmax = fmaxf(lmax, v);
          lmin = fminf(lmin, v);
        } else {
          // row = n, col = d -> out[b][c][4i+mr][4j+mc]
          const int n = row, d = col;
          const int bb = n & 127, t2 = n >> 7, j = t2 % 14, i = t2 / 14;
          const int c = d >> 4, mr = (d >> 2) & 3, mc = d & 3;
          ((float*)outp)[((bb * CC + c) * HH + i * 4 + mr) * HH + j * 4 + mc] = v;
        }
      }
    }
  }

  if (MODE == 1) {
    __shared__ float smax[8], smin[8];
    for (int off = 16; off; off >>= 1) {
      lmax = fmaxf(lmax, __shfl_xor(lmax, off));
      lmin = fminf(lmin, __shfl_xor(lmin, off));
    }
    if (lane == 0) { smax[wid] = lmax; smin[wid] = lmin; }
    __syncthreads();
    if (threadIdx.x == 0) {
      float M = smax[0], m = smin[0];
      for (int w = 1; w < 8; ++w) { M = fmaxf(M, smax[w]); m = fminf(m, smin[w]); }
      atomicMax(&mm[0], monoKey(M));
      atomicMax(&mm[1], monoKey(-m));
    }
  }
}

// -------------------------------------------------- Jacobi eigensolver ------
__global__ void copyWV_kernel(const float* __restrict__ cov, float* __restrict__ W,
                              float* __restrict__ V) {
  const int t = blockIdx.x * blockDim.x + threadIdx.x;
  W[t] = cov[t];                              // symmetric: row/col major identical
  V[t] = ((t >> 10) == (t & (DD - 1))) ? 1.0f : 0.0f;
}

// One-sided Jacobi: 512 disjoint column pairs (round-robin), rotate W and V cols.
__global__ void jacobi_round(float* __restrict__ W, float* __restrict__ V, int r) {
  __shared__ float red[3][8];
  __shared__ float cs[2];
  const int m = DD - 1;  // 1023
  const int k = blockIdx.x;
  int p = (r + k) % m;
  int q = (k == 0) ? m : (r + m - k) % m;
  if (p > q) { int t = p; p = q; q = t; }
  float* wp = W + (size_t)p * DD;
  float* wq = W + (size_t)q * DD;
  float* vp = V + (size_t)p * DD;
  float* vq = V + (size_t)q * DD;

  float app = 0.f, aqq = 0.f, apq = 0.f;
  for (int i = threadIdx.x; i < DD; i += blockDim.x) {
    const float a = wp[i], b = wq[i];
    app += a * a; aqq += b * b; apq += a * b;
  }
  for (int off = 16; off; off >>= 1) {
    app += __shfl_xor(app, off);
    aqq += __shfl_xor(aqq, off);
    apq += __shfl_xor(apq, off);
  }
  const int wid = threadIdx.x >> 5, lane = threadIdx.x & 31;
  if (!lane) { red[0][wid] = app; red[1][wid] = aqq; red[2][wid] = apq; }
  __syncthreads();
  if (threadIdx.x == 0) {
    float Ap = 0.f, Aq = 0.f, G = 0.f;
    const int nw = blockDim.x >> 5;
    for (int w = 0; w < nw; ++w) { Ap += red[0][w]; Aq += red[1][w]; G += red[2][w]; }
    float c = 1.f, s = 0.f;
    if (fabsf(G) > 1e-28f) {
      const float z = (Aq - Ap) / (2.f * G);
      const float t = copysignf(1.f, z) / (fabsf(z) + sqrtf(1.f + z * z));
      c = rsqrtf(1.f + t * t);
      s = t * c;
    }
    cs[0] = c; cs[1] = s;
  }
  __syncthreads();
  const float c = cs[0], s = cs[1];
  for (int i = threadIdx.x; i < DD; i += blockDim.x) {
    const float a = wp[i], b = wq[i];
    wp[i] = c * a - s * b;
    wq[i] = s * a + c * b;
    const float va = vp[i], vb = vq[i];
    vp[i] = c * va - s * vb;
    vq[i] = s * va + c * vb;
  }
}

// u = V (columns are eigenvectors). ut[m][k] = V[m*D+k]; u[d][k] = V[k*D+d].
__global__ void uprep_kernel(const float* __restrict__ V, __bf16* __restrict__ ub,
                             __bf16* __restrict__ utb) {
  const int t = blockIdx.x * blockDim.x + threadIdx.x;
  const int kk = t & (DD - 1), d = t >> 10;
  utb[t] = (__bf16)V[t];
  ub[t]  = (__bf16)V[(size_t)kk * DD + d];
}

// ------------------------------------------------------------- quantize -----
__global__ void quant_kernel(__bf16* __restrict__ P, const unsigned* __restrict__ mm) {
  const float maxv = monoDec(mm[0]);
  const float minv = -monoDec(mm[1]);
  const float scale = 255.0f / (maxv - minv);
  const float inv = 1.0f / scale;
  const int t = blockIdx.x * blockDim.x + threadIdx.x;
  const float v = (float)P[t];
  P[t] = (__bf16)(rintf(v * scale) * inv);
}

// ---------------------------------------------------------------------------
extern "C" void kernel_launch(void* const* d_in, const int* in_sizes, int n_in,
                              void* d_out, int out_size, void* d_ws, size_t ws_size,
                              hipStream_t stream) {
  const float* x = (const float*)d_in[0];
  float* out = (float*)d_out;

  char* ws = (char*)d_ws;
  size_t off = 0;
  auto carve = [&](size_t bytes) { char* p = ws + off; off += (bytes + 255) & ~(size_t)255; return p; };

  float*   mean = (float*)carve((size_t)DD * 4);
  __bf16*  im   = (__bf16*)carve((size_t)DD * NN * 2);
  __bf16*  imT  = (__bf16*)carve((size_t)DD * NN * 2);
  float*   cov  = (float*)carve((size_t)DD * DD * 4);
  float*   W    = (float*)carve((size_t)DD * DD * 4);
  float*   V    = (float*)carve((size_t)DD * DD * 4);
  __bf16*  ub   = (__bf16*)carve((size_t)DD * DD * 2);
  __bf16*  utb  = (__bf16*)carve((size_t)DD * DD * 2);
  __bf16*  P    = (__bf16*)carve((size_t)NN * DD * 2);
  unsigned* mm  = (unsigned*)carve(8);

  hipMemsetAsync(mm, 0, 8, stream);

  // 1) per-row mean
  mean_kernel<<<DD, 256, 0, stream>>>(x, mean);

  // 2) gather, mean-subtract, bf16 (both layouts)
  const int totalIm = DD * NN;  // 25,690,112 = 256 * 100352
  gather_im<<<totalIm / 256, 256, 0, stream>>>(x, mean, im);
  gather_imT<<<totalIm / 256, 256, 0, stream>>>(x, mean, imT);

  // 3) cov = im * im^T / N   [1024 x 1024], K = 25088
  gemm_wmma<0><<<dim3(DD / 128, DD / 128), 256, 0, stream>>>(
      im, im, NN, NN, NN, cov, DD, mm);

  // 4) eigenvectors of cov via one-sided Jacobi (V accumulates rotations)
  copyWV_kernel<<<(DD * DD) / 256, 256, 0, stream>>>(cov, W, V);
  for (int s = 0; s < SWEEPS; ++s)
    for (int r = 0; r < DD - 1; ++r)
      jacobi_round<<<DD / 2, 256, 0, stream>>>(W, V, r);
  uprep_kernel<<<(DD * DD) / 256, 256, 0, stream>>>(V, ub, utb);

  // 5) P = imProj^T = im^T * u   [25088 x 1024], K = 1024  (+ global max/min)
  gemm_wmma<1><<<dim3(DD / 128, NN / 128), 256, 0, stream>>>(
      imT, utb, DD, DD, DD, P, DD, mm);

  // 6) quantize in place (round-to-even, global scale)
  quant_kernel<<<(NN * DD) / 256, 256, 0, stream>>>(P, mm);

  // 7) back^T = Q * u^T  -> scatter to NCHW output
  gemm_wmma<2><<<dim3(DD / 128, NN / 128), 256, 0, stream>>>(
      P, ub, DD, DD, DD, out, DD, mm);
}
